// FastKNN_78469052498161
// MI455X (gfx1250) — matrix operand order repaired and checked by reading, not compiled, not run on hardware
//
#include <hip/hip_runtime.h>
#include <hip/hip_bf16.h>

typedef __attribute__((ext_vector_type(16))) __bf16 v16bf;
typedef __attribute__((ext_vector_type(8)))  float  v8f;
typedef __attribute__((ext_vector_type(4)))  unsigned int u32x4;
typedef __attribute__((ext_vector_type(8)))  int  i32x8;
typedef __attribute__((ext_vector_type(4)))  int  i32x4;

#define LSEQ  2048
#define BATCH 8
#define NH    1024
#define MROWS (LSEQ * BATCH)   // 16384
#define KTILE 32
#define MTILE 128
#define NTILE 32

#if defined(__has_builtin)
#if __has_builtin(__builtin_amdgcn_tensor_load_to_lds)
#define HAVE_TDM 1
#endif
#endif

__device__ __forceinline__ unsigned short f2bf(float f) {
    unsigned int u = __float_as_uint(f);
    u += 0x7FFFu + ((u >> 16) & 1u);
    return (unsigned short)(u >> 16);
}
__device__ __forceinline__ float sigmoidf_(float x) {
    return 1.0f / (1.0f + __expf(-x));
}

union FragAB { v16bf v; float4 f4[2]; };
union Acc    { v8f   v; float  f[8]; };

// ---------------- fp32 -> bf16 conversion ----------------
__global__ void k_cvt_bf16(const float* __restrict__ src,
                           unsigned short* __restrict__ dst, int n) {
    int i = blockIdx.x * blockDim.x + threadIdx.x;
    if (i < n) dst[i] = f2bf(src[i]);
}

// Stage a 128x32 bf16 A tile into LDS. All descriptor math is uniform (scalar);
// only the TDM issue is predicated on thread 0.
__device__ __forceinline__ void stage_tile(unsigned short* dstLds,
                                           const unsigned short* src, // xb + blockM*NH + kk
                                           int tid) {
#ifdef HAVE_TDM
    const unsigned long long ga  = (unsigned long long)src;
    const unsigned int       lds = (unsigned int)(unsigned long long)dstLds;
    u32x4 g0 = {0, 0, 0, 0};
    g0[0] = 1u;                                    // count=1 (valid user D#)
    g0[1] = lds;                                   // lds_addr
    g0[2] = (unsigned int)(ga & 0xffffffffu);      // global_addr[31:0]
    g0[3] = (unsigned int)((ga >> 32) & 0x01ffffffu) | (2u << 30); // addr[56:32]|type=2
    i32x8 g1 = {0, 0, 0, 0, 0, 0, 0, 0};
    g1[0] = (1 << 16);                             // data_size=1 (2B), wg_mask=0
    g1[1] = (int)((NH & 0xffff) << 16);            // tensor_dim0[15:0]
    g1[2] = (int)(((NH >> 16) & 0xffff) | ((MROWS & 0xffff) << 16));
    g1[3] = (int)(((MROWS >> 16) & 0xffff) | (KTILE << 16));   // tile_dim0
    g1[4] = MTILE;                                 // tile_dim1 (tile_dim2=0)
    g1[5] = NH;                                    // tensor_dim0_stride[31:0]
    i32x4 z4 = {0, 0, 0, 0};
    if (tid == 0) {
#if __clang_major__ >= 23
        i32x8 z8 = {0, 0, 0, 0, 0, 0, 0, 0};
        __builtin_amdgcn_tensor_load_to_lds(g0, g1, z4, z4, z8, 0);
#else
        __builtin_amdgcn_tensor_load_to_lds(g0, g1, z4, z4, 0);
#endif
    }
#else
    // cooperative fallback: 256 threads x 16 ushort = 4096 elements
    int t = tid * 16;
    int r = t >> 5;            // row in tile (cols = 32)
    int c = t & 31;            // col in tile (0 or 16)
    ((float4*)(dstLds + t))[0] = *(const float4*)(src + (size_t)r * NH + c);
    ((float4*)(dstLds + t))[1] = *(const float4*)(src + (size_t)r * NH + c + 8);
#endif
}

// ---------------- fused 3-gate GEMM + activation epilogue ----------------
// out[m,n] = sum_k x[m,k]*W[n,k] for W in {Wi,Wl,Wh}. Block: 128(M) x 32(N).
// Wave: 32(M) x 16(N) x 3 gates = 6 accumulators, 6 WMMAs per 32-K step.
__global__ __launch_bounds__(256) void k_gate_gemm(
    const unsigned short* __restrict__ xb,     // [MROWS][NH] bf16
    const unsigned short* __restrict__ wcat,   // [3][NH][NH] bf16
    const float* __restrict__ bl, const float* __restrict__ bh,
    float* __restrict__ wPlane, float* __restrict__ dPlane,
    float* __restrict__ tPlane) {
    __shared__ unsigned short tileA[2][MTILE * KTILE];   // 2 x 8 KB

    const int K = NH;
    const int tid = threadIdx.x;
    const int wave = tid >> 5, lane = tid & 31;
    const int laneLo = lane & 15, laneHi = lane >> 4;
    const int ni = wave & 1, mi = wave >> 1;             // 4 (M) x 2 (N) waves
    const int blockM = blockIdx.x * MTILE;
    const int mBase = blockM + mi * 32;
    const int nBase = blockIdx.y * NTILE + ni * 16;
    const unsigned short* aBase = xb + (size_t)blockM * NH;  // uniform tile origin

    Acc acc[2][3];
    #pragma unroll
    for (int s = 0; s < 2; ++s)
        #pragma unroll
        for (int g = 0; g < 3; ++g)
            #pragma unroll
            for (int r = 0; r < 8; ++r) acc[s][g].f[r] = 0.f;

    // B fragment row pointers (B[k][n] = W[n][k], contiguous in k)
    const unsigned short* bRow[3];
    #pragma unroll
    for (int g = 0; g < 3; ++g)
        bRow[g] = wcat + (size_t)(g * NH + nBase + laneLo) * K;

    // per-wave LDS fragment offsets (static per buffer)
    const int ldsOff0 = (mi * 32 + laneLo) * KTILE + laneHi * 8;        // frag s=0
    const int ldsOff1 = (mi * 32 + 16 + laneLo) * KTILE + laneHi * 8;   // frag s=1

    // one 32-K step from LDS buffer `lA` against K-chunk index i.
    // All 5 fragments (2xA from LDS, 3xB from global) are issued before the
    // 6-WMMA chain so the scheduler can use partial loadcnt/dscnt waits.
    auto kstep = [&](const unsigned short* lA, int i) {
        FragAB fa0, fa1, fb[3];
        fa0.f4[0] = *(const float4*)(lA + ldsOff0);
        fa0.f4[1] = *(const float4*)(lA + ldsOff0 + 16);
        fa1.f4[0] = *(const float4*)(lA + ldsOff1);
        fa1.f4[1] = *(const float4*)(lA + ldsOff1 + 16);
        const int kb = i * KTILE + laneHi * 8;
        #pragma unroll
        for (int g = 0; g < 3; ++g) {
            fb[g].f4[0] = *(const float4*)(bRow[g] + kb);
            fb[g].f4[1] = *(const float4*)(bRow[g] + kb + 16);
        }
        #pragma unroll
        for (int g = 0; g < 3; ++g) {
            acc[0][g].v = __builtin_amdgcn_wmma_f32_16x16x32_bf16(
                false, fa0.v, false, fb[g].v, (short)0, acc[0][g].v, false, false);
            acc[1][g].v = __builtin_amdgcn_wmma_f32_16x16x32_bf16(
                false, fa1.v, false, fb[g].v, (short)0, acc[1][g].v, false, false);
        }
    };

    const int nChunks = K / KTILE;   // 32 (even)

    stage_tile(tileA[0], aBase, tid);                    // prologue: chunk 0 -> buf0
    for (int i = 0; i < nChunks; i += 2) {
        // ---- chunk i from buf0; DMA chunk i+1 -> buf1 under compute ----
#ifdef HAVE_TDM
        if (tid == 0) __builtin_amdgcn_s_wait_tensorcnt(0);
#endif
        __syncthreads();
        if (i + 1 < nChunks)
            stage_tile(tileA[1], aBase + (i + 1) * KTILE, tid);
        kstep(tileA[0], i);

        // ---- chunk i+1 from buf1; DMA chunk i+2 -> buf0 under compute ----
#ifdef HAVE_TDM
        if (tid == 0) __builtin_amdgcn_s_wait_tensorcnt(0);
#endif
        __syncthreads();
        if (i + 2 < nChunks)
            stage_tile(tileA[0], aBase + (i + 2) * KTILE, tid);
        kstep(tileA[1], i + 1);
    }

    // D layout: VGPR r -> (M = base + r + (lane>=16 ? 8:0), N = nBase + lane%16)
    const int n = nBase + laneLo;
    const float blv = bl[n], bhv = bh[n];
    #pragma unroll
    for (int s = 0; s < 2; ++s)
        #pragma unroll
        for (int r = 0; r < 8; ++r) {
            const int m = mBase + s * 16 + r + laneHi * 8;
            const size_t o = (size_t)m * NH + n;
            const float dec = sigmoidf_(acc[s][1].f[r] + blv);
            const float tt  = sigmoidf_(acc[s][2].f[r] + bhv);
            wPlane[o] = acc[s][0].f[r] * (1.0f - dec);
            dPlane[o] = dec;
            tPlane[o] = tt;
        }
}

// ---------------- sequential scan: c = d*c + w, in-place into wPlane ----------------
__global__ void k_scan(float* __restrict__ wc, const float* __restrict__ d,
                       const float* __restrict__ c0) {
    const int j = blockIdx.x * blockDim.x + threadIdx.x;  // 0..8191 == b*NH + o
    float c = c0[j];
    size_t idx = (size_t)j;
    for (int l = 0; l < LSEQ; ++l) {
        c = d[idx] * c + wc[idx];
        wc[idx] = c;
        idx += (size_t)BATCH * NH;
    }
}

// ---------------- h = c*t + x*(1-t); next-layer bf16 input + last-step h/c ----------------
__global__ void k_combine(const float* __restrict__ c, const float* __restrict__ t,
                          const float* __restrict__ x, float* __restrict__ hOut,
                          unsigned short* __restrict__ xbNext,
                          float* __restrict__ hLast, float* __restrict__ cLast) {
    const size_t i = (size_t)blockIdx.x * blockDim.x + threadIdx.x;
    if (i >= (size_t)MROWS * NH) return;
    const float tt = t[i];
    const float cv = c[i];
    const float h  = cv * tt + x[i] * (1.0f - tt);
    hOut[i] = h;
    if (xbNext) xbNext[i] = f2bf(h);
    const int m = (int)(i >> 10);
    if (m >= MROWS - BATCH) {
        const int b = m - (MROWS - BATCH);
        const int o = (int)(i & (NH - 1));
        hLast[b * NH + o] = h;
        cLast[b * NH + o] = cv;
    }
}

extern "C" void kernel_launch(void* const* d_in, const int* in_sizes, int n_in,
                              void* d_out, int out_size, void* d_ws, size_t ws_size,
                              hipStream_t stream) {
    (void)in_sizes; (void)n_in; (void)out_size; (void)ws_size;
    // input order: input, h0, c0, Wi0, Wl0, bl0, Wh0, bh0, Wi1, Wl1, bl1, Wh1, bh1
    const float* x_in = (const float*)d_in[0];
    const float* c0   = (const float*)d_in[2];
    const float* Ws[6] = { (const float*)d_in[3], (const float*)d_in[4], (const float*)d_in[6],
                           (const float*)d_in[8], (const float*)d_in[9], (const float*)d_in[11] };
    const float* bls[2] = { (const float*)d_in[5],  (const float*)d_in[10] };
    const float* bhs[2] = { (const float*)d_in[7],  (const float*)d_in[12] };

    char* ws = (char*)d_ws;
    size_t off = 0;
    unsigned short* Wcat = (unsigned short*)(ws + off); off += (size_t)2 * 3 * NH * NH * 2;
    unsigned short* xb   = (unsigned short*)(ws + off); off += (size_t)MROWS * NH * 2;
    float* wPlane = (float*)(ws + off); off += (size_t)MROWS * NH * 4;
    float* dPlane = (float*)(ws + off); off += (size_t)MROWS * NH * 4;
    float* tPlane = (float*)(ws + off); off += (size_t)MROWS * NH * 4;
    float* xnext  = (float*)(ws + off); off += (size_t)MROWS * NH * 4;

    float* outX = (float*)d_out;
    float* outH = outX + (size_t)MROWS * NH;
    float* outC = outH + (size_t)2 * BATCH * NH;

    for (int li = 0; li < 2; ++li)
        for (int g = 0; g < 3; ++g) {
            const int n = NH * NH;
            k_cvt_bf16<<<(n + 255) / 256, 256, 0, stream>>>(
                Ws[li * 3 + g], Wcat + ((size_t)(li * 3 + g)) * NH * NH, n);
        }
    {
        const int n = MROWS * NH;
        k_cvt_bf16<<<(n + 255) / 256, 256, 0, stream>>>(x_in, xb, n);
    }

    const float* xcur = x_in;
    for (int li = 0; li < 2; ++li) {
        dim3 grid(MROWS / MTILE, NH / NTILE);
        k_gate_gemm<<<grid, 256, 0, stream>>>(
            xb, Wcat + (size_t)li * 3 * NH * NH, bls[li], bhs[li],
            wPlane, dPlane, tPlane);

        k_scan<<<(BATCH * NH) / 256, 256, 0, stream>>>(
            wPlane, dPlane, c0 + (size_t)li * BATCH * NH);

        float* hOut = (li == 1) ? outX : xnext;
        unsigned short* xbN = (li == 0) ? xb : nullptr;
        const size_t tot = (size_t)MROWS * NH;
        k_combine<<<(unsigned)((tot + 255) / 256), 256, 0, stream>>>(
            wPlane, tPlane, xcur, hOut, xbN,
            outH + (size_t)li * BATCH * NH, outC + (size_t)li * BATCH * NH);
        xcur = xnext;
    }
}
